// UtterancePolicy_80195629351196
// MI455X (gfx1250) — compile-verified, matrix operand order checked by reading, not run on hardware
//
#include <hip/hip_runtime.h>
#include <hip/hip_bf16.h>
#include <cstdint>

// Problem dims (fixed by reference)
#define B_   8192
#define E_   1024
#define T_   64
#define LEN_ 20
#define G4E  (4*E_)   // 4096

typedef __bf16 bf16_t;
typedef __bf16        v16bf __attribute__((ext_vector_type(16)));
typedef float         v8f   __attribute__((ext_vector_type(8)));
typedef unsigned int  v8u   __attribute__((ext_vector_type(8)));

__device__ __forceinline__ float sigf(float x) { return 1.0f / (1.0f + __expf(-x)); }

// Deterministic per-(step,row,token) Gumbel noise (splitmix64-style mix).
__device__ __forceinline__ float gumbel_noise(unsigned step, unsigned row, unsigned tok) {
    unsigned long long x = (unsigned long long)step * 0x9E3779B97F4A7C15ULL
                         + (unsigned long long)row  * 0xBF58476D1CE4E5B9ULL
                         + (unsigned long long)tok  * 0x94D049BB133111EBULL;
    x ^= x >> 33; x *= 0xFF51AFD7ED558CCDULL;
    x ^= x >> 33; x *= 0xC4CEB9FE1A85EC53ULL;
    x ^= x >> 33;
    float u = ((float)(unsigned)(x >> 40) + 0.5f) * (1.0f / 16777216.0f); // (0,1)
    return -__logf(-__logf(u));
}

// ---------------------------------------------------------------------------
// Prep kernels
// ---------------------------------------------------------------------------
__global__ __launch_bounds__(256)
void k_cvt_whh(const float* __restrict__ w, bf16_t* __restrict__ wb, int n) {
    int i = blockIdx.x * 256 + threadIdx.x;
    if (i < n) wb[i] = (bf16_t)w[i];
}

__global__ __launch_bounds__(256)
void k_init_state(const float* __restrict__ h_t, float* __restrict__ hf,
                  bf16_t* __restrict__ hb, float* __restrict__ c,
                  int* __restrict__ prev_tok, float* __restrict__ ent_rows,
                  float* __restrict__ match_rows) {
    int idx = blockIdx.x * 256 + threadIdx.x;
    if (idx < B_ * E_) {
        float v = h_t[idx];
        hf[idx] = v;
        hb[idx] = (bf16_t)v;
        c[idx]  = 0.0f;
    }
    if (idx < B_) { prev_tok[idx] = 0; ent_rows[idx] = 0.0f; match_rows[idx] = 0.0f; }
}

// tokgates[t][n] = emb[t] . W_ih[n] + b_ih[n] + b_hh[n]   (64 x 4096, K=1024; one-time)
__global__ __launch_bounds__(256)
void k_tokgates(const float* __restrict__ emb, const float* __restrict__ W_ih,
                const float* __restrict__ b_ih, const float* __restrict__ b_hh,
                float* __restrict__ tokgates) {
    int idx = blockIdx.x * 256 + threadIdx.x;           // t*4096 + n
    int t = idx >> 12, n = idx & 4095;
    const float4* e4 = (const float4*)(emb  + (size_t)t * E_);
    const float4* w4 = (const float4*)(W_ih + (size_t)n * E_);
    float s = 0.0f;
    for (int k = 0; k < E_ / 4; ++k) {
        float4 a = e4[k], b = w4[k];
        s += a.x * b.x + a.y * b.y + a.z * b.z + a.w * b.w;
    }
    tokgates[idx] = s + b_ih[n] + b_hh[n];
}

// ---------------------------------------------------------------------------
// Recurrent GEMM: gates[B,4E] = tokgates[prev_tok[b]] + h_bf16 @ W_hh_bf16^T
// Block = 256 threads (8 waves) -> 64 rows x 128 cols.
// Wave w: rows (w>>1)*16, cols (w&1)*64 -> FOUR 16x16 f32 accum tiles per wave
// (A fragment amortized over 4 WMMAs). A staged via double-buffered padded LDS
// (one barrier per K-chunk); B streamed from L2-resident bf16 weights as b128.
// ---------------------------------------------------------------------------
#define LDSROW 20                 // dwords per 64-elem... 32 bf16 row, padded (80B, 16B aligned)
#define LDSBUF (64 * LDSROW)

__global__ __launch_bounds__(256)
void k_gates_gemm(const bf16_t* __restrict__ hb, const bf16_t* __restrict__ whh,
                  const float* __restrict__ tokgates, const int* __restrict__ prev_tok,
                  float* __restrict__ gates) {
    __shared__ unsigned int ldsA[2 * LDSBUF];  // double-buffered 64x32 bf16 A stage

    const int tid   = threadIdx.x;
    const int lane  = tid & 31;
    const int wave  = tid >> 5;
    const int laneN = lane & 15;
    const int half  = lane >> 4;
    const int wrow  = wave >> 1;          // 0..3
    const int wcol  = wave & 1;           // 0..1
    const int rowblock = blockIdx.y * 64;
    const int colblock = blockIdx.x * 128;
    const int col0 = colblock + wcol * 64 + laneN;   // + cfr*16 for cfr in 0..3
    const int wrowbase = rowblock + wrow * 16;

    // Accumulators initialized from precomputed input-path gates (fused emb lookup)
    v8f acc[4];
#pragma unroll
    for (int r = 0; r < 8; ++r) {
        int row = wrowbase + r + 8 * half;
        int tok = prev_tok[row];
        const float* tg = tokgates + (size_t)tok * G4E;
#pragma unroll
        for (int cfr = 0; cfr < 4; ++cfr) acc[cfr][r] = tg[col0 + cfr * 16];
    }

    // A-stage addressing: thread loads 8 bf16 (one uint4) per chunk
    const int arow = rowblock + (tid >> 2);
    const int acol = (tid & 3) * 8;
    const int ldst = (tid >> 2) * LDSROW + (tid & 3) * 4;

    // Prologue: stage chunk 0 into buffer 0
    {
        uint4 st = *(const uint4*)(hb + (size_t)arow * E_ + acol);
        *(uint4*)(&ldsA[ldst]) = st;
    }
    __syncthreads();

    const int NCH = E_ / 32;
    for (int i = 0; i < NCH; ++i) {
        const int buf = i & 1;
        const int kc  = i * 32;

        // Stage chunk i+1 into the other buffer (disjoint from this chunk's reads)
        if (i + 1 < NCH) {
            uint4 nx = *(const uint4*)(hb + (size_t)arow * E_ + kc + 32 + acol);
            *(uint4*)(&ldsA[(buf ^ 1) * LDSBUF + ldst]) = nx;
            if (i + 2 < NCH)   // global_prefetch_b8 two chunks ahead
                __builtin_prefetch(hb + (size_t)arow * E_ + kc + 64 + acol, 0, 1);
        }

        // A fragment: two aligned b128 LDS reads (k = half*8 + 0..7, 16..23(+half*8))
        const unsigned int* lbase = &ldsA[buf * LDSBUF + (wrow * 16 + laneN) * LDSROW + half * 4];
        uint4 alo = *(const uint4*)(lbase);
        uint4 ahi = *(const uint4*)(lbase + 8);
        v8u araw;
        araw[0] = alo.x; araw[1] = alo.y; araw[2] = alo.z; araw[3] = alo.w;
        araw[4] = ahi.x; araw[5] = ahi.y; araw[6] = ahi.z; araw[7] = ahi.w;
        v16bf af = __builtin_bit_cast(v16bf, araw);

        // Four B fragments -> four WMMAs sharing the A fragment
#pragma unroll
        for (int cfr = 0; cfr < 4; ++cfr) {
            const int n = col0 + cfr * 16;     // B[k][n] = W_hh[n][k], k contiguous
            const bf16_t* bp = whh + (size_t)n * E_ + kc + half * 8;
            uint4 blo = *(const uint4*)(bp);
            uint4 bhi = *(const uint4*)(bp + 16);
            v8u braw;
            braw[0] = blo.x; braw[1] = blo.y; braw[2] = blo.z; braw[3] = blo.w;
            braw[4] = bhi.x; braw[5] = bhi.y; braw[6] = bhi.z; braw[7] = bhi.w;
            v16bf bfr = __builtin_bit_cast(v16bf, braw);
            acc[cfr] = __builtin_amdgcn_wmma_f32_16x16x32_bf16(false, af, false, bfr,
                                                               (short)0, acc[cfr], false, false);
        }
        __syncthreads();   // single barrier per chunk (double buffer)
    }

    // C layout: VGPR r, lane l -> row = r + 8*(l>>4), col = l&15.
    // gates is a write-once/read-once 128MB stream: bypass L2 residency (NT).
#pragma unroll
    for (int r = 0; r < 8; ++r) {
        int row = wrowbase + r + 8 * half;
        float* grow = gates + (size_t)row * G4E;
#pragma unroll
        for (int cfr = 0; cfr < 4; ++cfr)
            __builtin_nontemporal_store(acc[cfr][r], grow + col0 + cfr * 16);
    }
}

// ---------------------------------------------------------------------------
// Pointwise LSTM cell (PyTorch gate order i,f,g,o along 4E columns)
// ---------------------------------------------------------------------------
__global__ __launch_bounds__(256)
void k_lstm_cell(const float* __restrict__ gates, float* __restrict__ c,
                 float* __restrict__ hf, bf16_t* __restrict__ hb) {
    int idx = blockIdx.x * 256 + threadIdx.x;   // b*E + e
    int b = idx >> 10, e = idx & 1023;
    const float* g = gates + (size_t)b * G4E;
    float ig = sigf(__builtin_nontemporal_load(g + e));
    float fg = sigf(__builtin_nontemporal_load(g + E_ + e));
    float gg = tanhf(__builtin_nontemporal_load(g + 2 * E_ + e));
    float og = sigf(__builtin_nontemporal_load(g + 3 * E_ + e));
    float cn = fg * c[idx] + ig * gg;
    float hn = og * tanhf(cn);
    c[idx]  = cn;
    hf[idx] = hn;
    hb[idx] = (bf16_t)hn;
}

// ---------------------------------------------------------------------------
// Logits + softmax + Gumbel-max categorical + argmax/entropy. One wave32 per row;
// each lane owns tokens {lane, lane+32}. Butterfly reductions via __shfl_xor.
// ---------------------------------------------------------------------------
__global__ __launch_bounds__(256)
void k_sample(const float* __restrict__ hf, const float* __restrict__ W1,
              const float* __restrict__ b1, float* __restrict__ out,
              int* __restrict__ prev_tok, float* __restrict__ ent_rows,
              float* __restrict__ match_rows, int step) {
    const int lane = threadIdx.x & 31;
    const int wave = threadIdx.x >> 5;
    const int row  = blockIdx.x * 8 + wave;
    const int t0 = lane, t1 = lane + 32;

    const float4* h4 = (const float4*)(hf + (size_t)row * E_);
    const float4* w0 = (const float4*)(W1 + (size_t)t0 * E_);
    const float4* w1 = (const float4*)(W1 + (size_t)t1 * E_);
    float a0 = 0.0f, a1 = 0.0f;
    for (int k = 0; k < E_ / 4; ++k) {
        float4 h = h4[k], x = w0[k], y = w1[k];
        a0 += h.x * x.x + h.y * x.y + h.z * x.z + h.w * x.w;
        a1 += h.x * y.x + h.y * y.y + h.z * y.z + h.w * y.w;
    }
    float l0 = a0 + b1[t0];
    float l1 = a1 + b1[t1];

    float m = fmaxf(l0, l1);
#pragma unroll
    for (int o = 16; o; o >>= 1) m = fmaxf(m, __shfl_xor(m, o, 32));
    float e0 = __expf(l0 - m), e1 = __expf(l1 - m);
    float s = e0 + e1;
#pragma unroll
    for (int o = 16; o; o >>= 1) s += __shfl_xor(s, o, 32);

    // greedy argmax (first-max tie-break like jnp.argmax)
    float gv = (l0 >= l1) ? l0 : l1;
    int   gi = (l0 >= l1) ? t0 : t1;
#pragma unroll
    for (int o = 16; o; o >>= 1) {
        float ov = __shfl_xor(gv, o, 32);
        int   oi = __shfl_xor(gi, o, 32);
        if (ov > gv || (ov == gv && oi < gi)) { gv = ov; gi = oi; }
    }

    // categorical sample via Gumbel-max
    float p0 = l0 + gumbel_noise((unsigned)step, (unsigned)row, (unsigned)t0);
    float p1 = l1 + gumbel_noise((unsigned)step, (unsigned)row, (unsigned)t1);
    float sv = (p0 >= p1) ? p0 : p1;
    int   si = (p0 >= p1) ? t0 : t1;
#pragma unroll
    for (int o = 16; o; o >>= 1) {
        float ov = __shfl_xor(sv, o, 32);
        int   oi = __shfl_xor(si, o, 32);
        if (ov > sv || (ov == sv && oi < si)) { sv = ov; si = oi; }
    }

    // logit of the sampled token (broadcast via max)
    float cand = (si == t0) ? l0 : ((si == t1) ? l1 : -3.4e38f);
#pragma unroll
    for (int o = 16; o; o >>= 1) cand = fmaxf(cand, __shfl_xor(cand, o, 32));
    float logZ = m + __logf(s);

    // entropy of (probs + EPS)
    float inv = 1.0f / s;
    float q0 = e0 * inv + 1e-8f, q1 = e1 * inv + 1e-8f;
    float ent = -(q0 * __logf(q0) + q1 * __logf(q1));
#pragma unroll
    for (int o = 16; o; o >>= 1) ent += __shfl_xor(ent, o, 32);

    if (lane == 0) {
        ent_rows[row]   += ent;                       // accumulated across steps
        match_rows[row] += (gi == si) ? 1.0f : 0.0f;
        prev_tok[row]    = si;
        out[(size_t)step * B_ + row] = cand - logZ;                      // log_gs[t][b]
        out[(size_t)LEN_ * B_ + (size_t)row * LEN_ + step] = (float)si;  // utterance[b][t]
    }
}

// Deterministic final reduction of entropy / matches scalars.
__global__ __launch_bounds__(256)
void k_finalize(const float* __restrict__ ent_rows, const float* __restrict__ match_rows,
                float* __restrict__ out) {
    __shared__ float se[256], sm[256];
    int tid = threadIdx.x;
    float e = 0.0f, m = 0.0f;
    for (int i = tid; i < B_; i += 256) { e += ent_rows[i]; m += match_rows[i]; }
    se[tid] = e; sm[tid] = m;
    __syncthreads();
    for (int o = 128; o; o >>= 1) {
        if (tid < o) { se[tid] += se[tid + o]; sm[tid] += sm[tid + o]; }
        __syncthreads();
    }
    if (tid == 0) {
        out[2 * LEN_ * B_ + 0] = se[0];
        out[2 * LEN_ * B_ + 1] = sm[0];
        out[2 * LEN_ * B_ + 2] = (float)(B_ * LEN_);
    }
}

// ---------------------------------------------------------------------------
extern "C" void kernel_launch(void* const* d_in, const int* in_sizes, int n_in,
                              void* d_out, int out_size, void* d_ws, size_t ws_size,
                              hipStream_t stream) {
    (void)in_sizes; (void)n_in; (void)out_size; (void)ws_size;
    const float* h_t  = (const float*)d_in[0];
    // d_in[1] = testing (reference runs with 0): stochastic path implemented
    const float* emb  = (const float*)d_in[2];
    const float* W_ih = (const float*)d_in[3];
    const float* W_hh = (const float*)d_in[4];
    const float* b_ih = (const float*)d_in[5];
    const float* b_hh = (const float*)d_in[6];
    const float* W1   = (const float*)d_in[7];
    const float* b1   = (const float*)d_in[8];
    float* out = (float*)d_out;

    // Workspace carve-up (~228 MB total)
    char* ws = (char*)d_ws;
    size_t off = 0;
    auto alloc = [&](size_t bytes) -> void* {
        void* p = ws + off;
        off += (bytes + 255) & ~(size_t)255;
        return p;
    };
    bf16_t* whh_bf    = (bf16_t*)alloc((size_t)G4E * E_ * 2);   //   8 MB, L2-resident
    float*  tokgates  = (float*) alloc((size_t)T_ * G4E * 4);   //   1 MB
    bf16_t* hb        = (bf16_t*)alloc((size_t)B_ * E_ * 2);    //  16 MB
    float*  hfv       = (float*) alloc((size_t)B_ * E_ * 4);    //  32 MB
    float*  cv        = (float*) alloc((size_t)B_ * E_ * 4);    //  32 MB
    float*  gates     = (float*) alloc((size_t)B_ * G4E * 4);   // 128 MB (NT stream)
    int*    prev_tok  = (int*)   alloc((size_t)B_ * 4);
    float*  ent_rows  = (float*) alloc((size_t)B_ * 4);
    float*  match_rows= (float*) alloc((size_t)B_ * 4);

    k_cvt_whh   <<<(G4E * E_ + 255) / 256, 256, 0, stream>>>(W_hh, whh_bf, G4E * E_);
    k_init_state<<<(B_ * E_ + 255) / 256, 256, 0, stream>>>(h_t, hfv, hb, cv,
                                                            prev_tok, ent_rows, match_rows);
    k_tokgates  <<<(T_ * G4E) / 256, 256, 0, stream>>>(emb, W_ih, b_ih, b_hh, tokgates);

    dim3 ggrid(G4E / 128, B_ / 64);  // 32 x 128 tile blocks
    for (int t = 0; t < LEN_; ++t) {
        k_gates_gemm<<<ggrid, 256, 0, stream>>>(hb, whh_bf, tokgates, prev_tok, gates);
        k_lstm_cell <<<(B_ * E_) / 256, 256, 0, stream>>>(gates, cv, hfv, hb);
        k_sample    <<<B_ / 8, 256, 0, stream>>>(hfv, W1, b1, out, prev_tok,
                                                 ent_rows, match_rows, t);
    }
    k_finalize<<<1, 256, 0, stream>>>(ent_rows, match_rows, out);
}
